// WindowFusion_438086664588
// MI455X (gfx1250) — matrix operand order repaired and checked by reading, not compiled, not run on hardware
//
#include <hip/hip_runtime.h>
#include <hip/hip_bf16.h>

// MI455X (gfx1250) fused window attention.
// Pipeline (all bandwidth-bound at 23.3 TB/s; bf16 WMMA w/ f32 accumulate):
//   1) q  = SCALE*(x @ Wq + bq)      -> bf16 scratch (52 MB)
//   2) kv = y @ Wkv + bkv            -> bf16 scratch (105 MB)
//   3) per-(window,head) attention w/ decomposed rel-pos bias + softmax -> f32 scratch
//   4) xo = attn @ Wproj + bproj     -> f32 (reuses dead kv region)
//   5) out1 = T(xo*x + y), out2 = T(sigmoid(xo))  (LDS-tiled transpose)
// Workspace requirement: 52,428,800 + 104,857,600 + 104,857,600 = 262,144,000 B.

typedef __attribute__((ext_vector_type(16))) __bf16 v16bf;
typedef __attribute__((ext_vector_type(8)))  __bf16 v8bf;
typedef __attribute__((ext_vector_type(8)))  float  v8f;
typedef __attribute__((ext_vector_type(4)))  float  v4f;

__device__ __forceinline__ v16bf cat8(v8bf lo, v8bf hi) {
  v16bf r;
#pragma unroll
  for (int i = 0; i < 8; ++i) { r[i] = lo[i]; r[i + 8] = hi[i]; }
  return r;
}

// ---------------------------------------------------------------------------
// Generic GEMM: C[M,N] = alpha * (A[M,K] @ W[K,N] + bias[N]).
// A,W fp32 in HBM, converted to bf16 while staging to LDS; f32 accumulate via
// v_wmma_f32_16x16x32_bf16. Block tile 128x128x32, 8 waves (wave32), each wave
// owns a 32x64 sub-tile = 2x4 WMMA accumulators.
// A-fragment (16x32, 16-bit): lane L -> row L%16, K group (L/16)*8 (+16 for
// VGPRs 4..7) => two contiguous 16B LDS loads per fragment.
// B-fragment (32x16): staged transposed ([N][K]) so each lane reads the
// same two 8-value K runs as A at its column => two 16B LDS loads.
// ---------------------------------------------------------------------------
template <typename OutT>
__global__ void __launch_bounds__(256) gemm_bias_wmma(
    const float* __restrict__ A, const float* __restrict__ W,
    const float* __restrict__ bias, OutT* __restrict__ C,
    int M, int N, int K, float alpha)
{
  constexpr int BK  = 32;
  constexpr int LDA = BK + 8;                 // 40 halves = 80 B row pitch
  __shared__ __bf16 As[128 * LDA];
  __shared__ __bf16 Bs[128 * LDA];            // transposed: [n][k]

  const int tid  = threadIdx.x;
  const int lane = tid & 31;
  const int wid  = tid >> 5;
  const int lcol = lane & 15;
  const int lhi  = lane >> 4;
  const int m0   = blockIdx.y * 128;
  const int n0   = blockIdx.x * 128;
  const int wm   = (wid >> 1) * 32;           // 4 waves down
  const int wn   = (wid & 1) * 64;            // 2 waves across

  v8f acc[2][4];
#pragma unroll
  for (int mt = 0; mt < 2; ++mt)
#pragma unroll
    for (int nt = 0; nt < 4; ++nt)
#pragma unroll
      for (int i = 0; i < 8; ++i) acc[mt][nt][i] = 0.f;

  const int arow = tid >> 1;                  // 0..127
  const int akq  = (tid & 1) * 16;            // 0 / 16
  const int brow = tid >> 3;                  // k 0..31
  const int bnq  = (tid & 7) * 16;            // n 0..112

  for (int k0 = 0; k0 < K; k0 += BK) {
    // --- stage A tile (fp32 -> bf16) ---
    {
      const float* ap = A + (size_t)(m0 + arow) * K + (k0 + akq);
      float fv[16];
#pragma unroll
      for (int i = 0; i < 4; ++i) {
        v4f t = *(const v4f*)(ap + i * 4);
#pragma unroll
        for (int j = 0; j < 4; ++j) fv[i * 4 + j] = t[j];
      }
      v8bf h0, h1;
#pragma unroll
      for (int i = 0; i < 8; ++i) { h0[i] = (__bf16)fv[i]; h1[i] = (__bf16)fv[8 + i]; }
      *(v8bf*)&As[arow * LDA + akq]     = h0;
      *(v8bf*)&As[arow * LDA + akq + 8] = h1;
    }
    // --- stage B tile transposed (fp32 -> bf16) ---
    {
      const float* wp = W + (size_t)(k0 + brow) * N + (n0 + bnq);
#pragma unroll
      for (int i = 0; i < 4; ++i) {
        v4f t = *(const v4f*)(wp + i * 4);
#pragma unroll
        for (int j = 0; j < 4; ++j)
          Bs[(bnq + i * 4 + j) * LDA + brow] = (__bf16)t[j];
      }
    }
    __syncthreads();

    v16bf af[2];
#pragma unroll
    for (int mt = 0; mt < 2; ++mt) {
      const int r  = wm + mt * 16 + lcol;
      const int kg = lhi * 8;
      af[mt] = cat8(*(const v8bf*)&As[r * LDA + kg],
                    *(const v8bf*)&As[r * LDA + kg + 16]);
    }
#pragma unroll
    for (int nt = 0; nt < 4; ++nt) {
      const int n  = wn + nt * 16 + lcol;
      const int kb = lhi * 8;
      v16bf bf = cat8(*(const v8bf*)&Bs[n * LDA + kb],
                      *(const v8bf*)&Bs[n * LDA + kb + 16]);
#pragma unroll
      for (int mt = 0; mt < 2; ++mt)
        acc[mt][nt] = __builtin_amdgcn_wmma_f32_16x16x32_bf16(
            false, af[mt], false, bf, (short)0, acc[mt][nt], false, false);
    }
    __syncthreads();
  }

  // epilogue: C layout (16x16 f32): VGPR r -> row (lhi*8 + r), lane -> col
#pragma unroll
  for (int nt = 0; nt < 4; ++nt) {
    const int col  = n0 + wn + nt * 16 + lcol;
    const float bv = bias[col];
#pragma unroll
    for (int mt = 0; mt < 2; ++mt) {
#pragma unroll
      for (int r = 0; r < 8; ++r) {
        const int row = m0 + wm + mt * 16 + lhi * 8 + r;
        C[(size_t)row * N + col] = (OutT)(alpha * (acc[mt][nt][r] + bv));
      }
    }
  }
}

// ---------------------------------------------------------------------------
// Attention: one workgroup per (window, head). Nw=100 tokens padded to 112
// queries (7 wave strips) and 128 keys (8 tiles, masked past 100).
// Static LDS (< 64 KB): Q/K region is reused for the bf16 prob matrix after
// the score WMMAs retire.
// ---------------------------------------------------------------------------
__global__ void __launch_bounds__(256) win_attn_wmma(
    const __bf16* __restrict__ Qg, const __bf16* __restrict__ KVg,
    const float* __restrict__ Rh, const float* __restrict__ Rw,
    float* __restrict__ Og)
{
  __shared__ __align__(16) char smem[59968];
  __bf16* Qs   = (__bf16*)smem;                // 112 x 72
  __bf16* Ks   = (__bf16*)(smem + 16128);      // 128 x 72
  __bf16* Ps   = (__bf16*)smem;                // 112 x 136 (aliases Qs/Ks)
  __bf16* Vt   = (__bf16*)(smem + 34560);      // 64 x 136 (V transposed [d][key])
  float*  relh = (float*)(smem + 51968);       // 100 x 10
  float*  relw = (float*)(smem + 55968);       // 100 x 10

  const int tid  = threadIdx.x;
  const int lane = tid & 31;
  const int wid  = tid >> 5;
  const int lcol = lane & 15;
  const int lhi  = lane >> 4;
  const int win  = blockIdx.x >> 3;
  const int hd   = blockIdx.x & 7;
  const int b    = win >> 6;
  const int wfl  = win & 63;
  const size_t tokbase = (size_t)b * 6400 + (size_t)(wfl >> 3) * 800 + (size_t)(wfl & 7) * 10;

  // ---- phase 1: stage Q (112 rows), K (128 rows), V transposed ----
  {
    const int row  = tid >> 1;    // 0..127
    const int half = tid & 1;     // 32-element halves of d=64
    const int iy = row / 10, ix = row - iy * 10;
    const size_t tok = tokbase + (size_t)iy * 80 + ix;

    if (row < 112) {
      v8bf* dst = (v8bf*)&Qs[row * 72 + half * 32];
      if (row < 100) {
        const __bf16* src = Qg + tok * 512 + hd * 64 + half * 32;
#pragma unroll
        for (int i = 0; i < 4; ++i) dst[i] = *(const v8bf*)(src + i * 8);
      } else {
        v8bf z;
#pragma unroll
        for (int i = 0; i < 8; ++i) z[i] = (__bf16)0.f;
#pragma unroll
        for (int i = 0; i < 4; ++i) dst[i] = z;
      }
    }
    {
      v8bf* dst = (v8bf*)&Ks[row * 72 + half * 32];
      if (row < 100) {
        const __bf16* src = KVg + tok * 1024 + hd * 64 + half * 32;
#pragma unroll
        for (int i = 0; i < 4; ++i) dst[i] = *(const v8bf*)(src + i * 8);
      } else {
        v8bf z;
#pragma unroll
        for (int i = 0; i < 8; ++i) z[i] = (__bf16)0.f;
#pragma unroll
        for (int i = 0; i < 4; ++i) dst[i] = z;
      }
    }
    if (row < 100) {
      const __bf16* src = KVg + tok * 1024 + 512 + hd * 64 + half * 32;
      v8bf tv[4];
#pragma unroll
      for (int i = 0; i < 4; ++i) tv[i] = *(const v8bf*)(src + i * 8);
#pragma unroll
      for (int i = 0; i < 32; ++i) Vt[(half * 32 + i) * 136 + row] = tv[i / 8][i % 8];
    } else {
#pragma unroll
      for (int i = 0; i < 32; ++i) Vt[(half * 32 + i) * 136 + row] = (__bf16)0.f;
    }
  }
  __syncthreads();

  // ---- decomposed rel-pos bias tables: relh[q][k]=q_row.Rh[qy-k+9], relw likewise
  for (int p = tid; p < 2000; p += 256) {
    const bool isw = p >= 1000;
    const int  pp  = isw ? p - 1000 : p;
    const int  q   = pp / 10;
    const int  j   = pp - q * 10;
    const int  coord = isw ? (q % 10) : (q / 10);
    const float* R = (isw ? Rw : Rh) + (size_t)(coord - j + 9) * 64;
    float s = 0.f;
#pragma unroll
    for (int d = 0; d < 64; ++d) s += (float)Qs[q * 72 + d] * R[d];
    (isw ? relw : relh)[q * 10 + j] = s;
  }

  // ---- scores: each active wave owns 16 query rows x 128 keys ----
  v8f s[8];
#pragma unroll
  for (int kt = 0; kt < 8; ++kt)
#pragma unroll
    for (int i = 0; i < 8; ++i) s[kt][i] = 0.f;

  if (wid < 7) {
    v16bf aq[2];
#pragma unroll
    for (int c = 0; c < 2; ++c) {
      const int r  = wid * 16 + lcol;
      const int kg = c * 32 + lhi * 8;
      aq[c] = cat8(*(const v8bf*)&Qs[r * 72 + kg],
                   *(const v8bf*)&Qs[r * 72 + kg + 16]);
    }
#pragma unroll
    for (int kt = 0; kt < 8; ++kt) {
      const int key = kt * 16 + lcol;
#pragma unroll
      for (int c = 0; c < 2; ++c) {
        const int kb = c * 32 + lhi * 8;
        v16bf bf = cat8(*(const v8bf*)&Ks[key * 72 + kb],
                        *(const v8bf*)&Ks[key * 72 + kb + 16]);
        s[kt] = __builtin_amdgcn_wmma_f32_16x16x32_bf16(
            false, aq[c], false, bf, (short)0, s[kt], false, false);
      }
    }
  }
  __syncthreads();   // all Qs/Ks reads retired before Ps overwrites the region

  // ---- bias + mask + softmax (registers + 16-lane shuffles), probs -> LDS bf16
  if (wid < 7) {
#pragma unroll
    for (int r = 0; r < 8; ++r) {
      const int  q   = wid * 16 + lhi * 8 + r;
      const bool qok = q < 100;
      float rv[8];
#pragma unroll
      for (int kt = 0; kt < 8; ++kt) {
        const int key = kt * 16 + lcol;
        float v = s[kt][r];
        if (qok && key < 100)
          v += relh[q * 10 + key / 10] + relw[q * 10 + key % 10];
        else
          v = -1e30f;
        rv[kt] = v;
      }
      float mx = rv[0];
#pragma unroll
      for (int kt = 1; kt < 8; ++kt) mx = fmaxf(mx, rv[kt]);
#pragma unroll
      for (int off = 1; off < 16; off <<= 1) mx = fmaxf(mx, __shfl_xor(mx, off, 32));
      float sum = 0.f;
#pragma unroll
      for (int kt = 0; kt < 8; ++kt) { rv[kt] = __expf(rv[kt] - mx); sum += rv[kt]; }
#pragma unroll
      for (int off = 1; off < 16; off <<= 1) sum += __shfl_xor(sum, off, 32);
      const float inv = 1.f / sum;
#pragma unroll
      for (int kt = 0; kt < 8; ++kt)
        Ps[q * 136 + kt * 16 + lcol] = (__bf16)(rv[kt] * inv);
    }
  }
  __syncthreads();

  // ---- out = P(112x128) @ V(128x64); store at original raster row (free window_reverse)
  if (wid < 7) {
    v8f o[4];
#pragma unroll
    for (int nt = 0; nt < 4; ++nt)
#pragma unroll
      for (int i = 0; i < 8; ++i) o[nt][i] = 0.f;

#pragma unroll
    for (int kc = 0; kc < 4; ++kc) {
      const int pr = wid * 16 + lcol;
      const int kg = kc * 32 + lhi * 8;
      v16bf ap = cat8(*(const v8bf*)&Ps[pr * 136 + kg],
                      *(const v8bf*)&Ps[pr * 136 + kg + 16]);
#pragma unroll
      for (int nt = 0; nt < 4; ++nt) {
        const int d  = nt * 16 + lcol;
        const int kb = kc * 32 + lhi * 8;
        v16bf bf = cat8(*(const v8bf*)&Vt[d * 136 + kb],
                        *(const v8bf*)&Vt[d * 136 + kb + 16]);
        o[nt] = __builtin_amdgcn_wmma_f32_16x16x32_bf16(
            false, ap, false, bf, (short)0, o[nt], false, false);
      }
    }
#pragma unroll
    for (int r = 0; r < 8; ++r) {
      const int q = wid * 16 + lhi * 8 + r;
      if (q < 100) {
        const int iy = q / 10, ix = q - iy * 10;
        float* dst = Og + (tokbase + (size_t)iy * 80 + ix) * 512 + hd * 64;
#pragma unroll
        for (int nt = 0; nt < 4; ++nt) dst[nt * 16 + lcol] = o[nt][r];
      }
    }
  }
}

// ---------------------------------------------------------------------------
// Fused epilogue with LDS-tiled transpose (B,HW,C) -> (B,C,HW):
//   out1 = xo*x + y ; out2 = sigmoid(xo). Coalesced reads AND writes.
// ---------------------------------------------------------------------------
__global__ void __launch_bounds__(256) fuse_out_transpose(
    const float* __restrict__ XO, const float* __restrict__ Xin,
    const float* __restrict__ Yin, float* __restrict__ out1,
    float* __restrict__ out2)
{
  __shared__ float t1[32][33];
  __shared__ float t2[32][33];
  const int b   = blockIdx.z;
  const int hw0 = blockIdx.x * 32;
  const int c0  = blockIdx.y * 32;
  const int j   = threadIdx.x & 31;
  const int i0  = threadIdx.x >> 5;

#pragma unroll
  for (int i = i0; i < 32; i += 8) {
    const size_t idx = ((size_t)b * 6400 + hw0 + i) * 512 + c0 + j;
    const float xo = XO[idx];
    t1[i][j] = xo * Xin[idx] + Yin[idx];
    t2[i][j] = 1.f / (1.f + __expf(-xo));
  }
  __syncthreads();
#pragma unroll
  for (int i = i0; i < 32; i += 8) {
    const size_t odx = ((size_t)b * 512 + c0 + i) * 6400 + hw0 + j;
    out1[odx] = t1[j][i];
    out2[odx] = t2[j][i];
  }
}

// ---------------------------------------------------------------------------
extern "C" void kernel_launch(void* const* d_in, const int* in_sizes, int n_in,
                              void* d_out, int out_size, void* d_ws, size_t ws_size,
                              hipStream_t stream)
{
  (void)in_sizes; (void)n_in; (void)out_size; (void)ws_size;

  const float* x     = (const float*)d_in[0];
  const float* y     = (const float*)d_in[1];
  const float* Wq    = (const float*)d_in[2];
  const float* bq    = (const float*)d_in[3];
  const float* Wkv   = (const float*)d_in[4];
  const float* bkv   = (const float*)d_in[5];
  const float* Wproj = (const float*)d_in[6];
  const float* bproj = (const float*)d_in[7];
  const float* rph   = (const float*)d_in[8];   // (19, 64)
  const float* rpw   = (const float*)d_in[9];

  float* out1 = (float*)d_out;
  float* out2 = out1 + (size_t)8 * 512 * 6400;

  char*   ws  = (char*)d_ws;
  __bf16* qb  = (__bf16*)ws;                                   // 52,428,800 B
  __bf16* kvb = (__bf16*)(ws + (size_t)52428800);              // 104,857,600 B
  float*  att = (float*)(ws + (size_t)52428800 + 104857600);   // 104,857,600 B
  float*  xo  = (float*)kvb;   // kv is dead after attention; reuse for xo (exact fit)

  const int   M     = 51200;    // 8 * 6400 tokens
  const float scale = 0.125f;   // HEAD_DIM^-0.5 folded into q

  gemm_bias_wmma<__bf16><<<dim3(4, 400), 256, 0, stream>>>(x, Wq, bq, qb, M, 512, 512, scale);
  gemm_bias_wmma<__bf16><<<dim3(8, 400), 256, 0, stream>>>(y, Wkv, bkv, kvb, M, 1024, 512, 1.0f);
  win_attn_wmma<<<dim3(4096), 256, 0, stream>>>(qb, kvb, rph, rpw, att);
  gemm_bias_wmma<float><<<dim3(4, 400), 256, 0, stream>>>(att, Wproj, bproj, xo, M, 512, 512, 1.0f);
  fuse_out_transpose<<<dim3(200, 16, 8), 256, 0, stream>>>(xo, x, y, out1, out2);
}
